// RelativeSelfMultiheadAttn_2233382993925
// MI455X (gfx1250) — matrix operand order repaired and checked by reading, not compiled, not run on hardware
//
#include <hip/hip_runtime.h>
#include <hip/hip_bf16.h>
#include <math.h>

// MI455X / gfx1250, wave32, WMMA f16->f32.
typedef __attribute__((ext_vector_type(16))) _Float16 v16h;
typedef __attribute__((ext_vector_type(4)))  _Float16 v4h;
typedef __attribute__((ext_vector_type(8)))  float    v8f;

constexpr int cT = 1024;   // sequence length
constexpr int cB = 4;      // batch
constexpr int cE = 1024;   // embed
constexpr int cH = 16;     // heads
constexpr int cD = 64;     // head dim
constexpr float cSCALE = 0.125f;  // 1/sqrt(64)

__device__ __forceinline__ v8f mma16(v16h a, v16h b, v8f c) {
  // v_wmma_f32_16x16x32_f16
  return __builtin_amdgcn_wmma_f32_16x16x32_f16(false, a, false, b, (short)0, c,
                                                false, false);
}

__device__ __forceinline__ v4h cvt4(float4 f) {
  v4h h;
  h[0] = (_Float16)f.x; h[1] = (_Float16)f.y;
  h[2] = (_Float16)f.z; h[3] = (_Float16)f.w;
  return h;
}

// A-matrix fragment (16x32 f16, M x K), row-major LDS base[row*stride + k]
__device__ __forceinline__ v16h frag_a(const _Float16* base, int stride, int m,
                                       int lane, int k0) {
  v16h a;
  const int hl = lane >> 4;
#pragma unroll
  for (int e = 0; e < 16; ++e) {
    int kk = k0 + ((e >> 3) << 4) + (hl << 3) + (e & 7);
    a[e] = base[m * stride + kk];
  }
  return a;
}

// B-matrix fragment (32x16 f16, K x N) where memory holds B^T: base[n*stride + k]
__device__ __forceinline__ v16h frag_bt(const _Float16* base, int stride, int n,
                                        int lane, int k0) {
  v16h b;
  const int hl = lane >> 4;
#pragma unroll
  for (int e = 0; e < 16; ++e) {
    int kk = k0 + ((e >> 3) << 4) + (hl << 3) + (e & 7);
    b[e] = base[n * stride + kk];
  }
  return b;
}

// ---------------- Bayesian weight sampling: out = mu + softplus(rho)*eps ----
// n is always a multiple of 4 here -> pure float4 kernel.
__global__ __launch_bounds__(256) void sample_gauss4(const float4* __restrict__ mu,
                                                     const float4* __restrict__ rho,
                                                     const float4* __restrict__ eps,
                                                     float4* __restrict__ out, int n4) {
  int i = blockIdx.x * 256 + threadIdx.x;
  if (i < n4) {
    float4 m = mu[i], r = rho[i], e = eps[i], o;
    o.x = m.x + log1pf(expf(r.x)) * e.x;
    o.y = m.y + log1pf(expf(r.y)) * e.y;
    o.z = m.z + log1pf(expf(r.z)) * e.z;
    o.w = m.w + log1pf(expf(r.w)) * e.w;
    out[i] = o;
  }
}

// ---------------- GEMM: C[M,N] = A[M,K] * W[N,K]^T + bias ------------------
// 64x128 tile per block (256 threads = 8 waves), K-step 32, f16 LDS staging.
// Each wave: 4 WMMA output tiles sharing one A fragment per K-step.
__global__ __launch_bounds__(256) void gemm_xwT(const float* __restrict__ A,
                                                const float* __restrict__ W,
                                                const float* __restrict__ bias,
                                                float* __restrict__ C,
                                                int M, int N, int K) {
  __shared__ _Float16 a_s[64 * 40];
  __shared__ _Float16 b_s[128 * 40];
  const int tid = threadIdx.x;
  const int lane = tid & 31, wv = tid >> 5;
  const int tm = wv & 3;              // 16-row tile index within 64
  const int grp = wv >> 2;            // 0/1 -> col tiles [0..3] / [4..7]
  const int m0 = blockIdx.y * 64, n0 = blockIdx.x * 128;
  v8f acc[4] = {{}, {}, {}, {}};

  for (int kt = 0; kt < K; kt += 32) {
    __syncthreads();
    // stage: A 64x32 (512 f4), W 128x32 (1024 f4); all loads in flight first
    float4 ra[2], rb[4];
#pragma unroll
    for (int i = 0; i < 2; ++i) {
      int idx4 = tid + i * 256;
      int r = idx4 >> 3, kk4 = (idx4 & 7) << 2;
      ra[i] = *(const float4*)&A[(size_t)(m0 + r) * K + kt + kk4];
    }
#pragma unroll
    for (int i = 0; i < 4; ++i) {
      int idx4 = tid + i * 256;
      int r = idx4 >> 3, kk4 = (idx4 & 7) << 2;
      rb[i] = *(const float4*)&W[(size_t)(n0 + r) * K + kt + kk4];
    }
    if (kt + 32 < K) {                // gfx1250 global_prefetch_b8, 1 line/row
      if (tid < 64)
        __builtin_prefetch(&A[(size_t)(m0 + tid) * K + kt + 32], 0, 3);
      else if (tid < 192)
        __builtin_prefetch(&W[(size_t)(n0 + tid - 64) * K + kt + 32], 0, 3);
    }
#pragma unroll
    for (int i = 0; i < 2; ++i) {
      int idx4 = tid + i * 256;
      int r = idx4 >> 3, kk4 = (idx4 & 7) << 2;
      *(v4h*)&a_s[r * 40 + kk4] = cvt4(ra[i]);
    }
#pragma unroll
    for (int i = 0; i < 4; ++i) {
      int idx4 = tid + i * 256;
      int r = idx4 >> 3, kk4 = (idx4 & 7) << 2;
      *(v4h*)&b_s[r * 40 + kk4] = cvt4(rb[i]);
    }
    __syncthreads();
    v16h a = frag_a(a_s, 40, 16 * tm + (lane & 15), lane, 0);
#pragma unroll
    for (int j = 0; j < 4; ++j) {
      v16h b = frag_bt(b_s, 40, 16 * (grp * 4 + j) + (lane & 15), lane, 0);
      acc[j] = mma16(a, b, acc[j]);
    }
  }

#pragma unroll
  for (int j = 0; j < 4; ++j) {
    int col = n0 + 16 * (grp * 4 + j) + (lane & 15);
    float bv = bias ? bias[col] : 0.f;
#pragma unroll
    for (int r = 0; r < 8; ++r) {
      int row = m0 + 16 * tm + ((lane >> 4) << 3) + r;
      C[(size_t)row * N + col] = acc[j][r] + bv;
    }
  }
}

// ---------------- Fused relative flash-attention ---------------------------
// grid = (T/64, B*H). Per 64-query tile: online softmax over causal key tiles.
// BD rel-shift handled by computing P = (q+rr)·Rsub^T over a 128-wide band and
// gathering the Toeplitz diagonal (col = 63-u+c) in LDS.
__global__ __launch_bounds__(256) void rel_attn(const float* __restrict__ qkv,
                                                const float* __restrict__ rbuf,
                                                const float* __restrict__ rw,
                                                const float* __restrict__ rr,
                                                float* __restrict__ ctx) {
  __shared__ _Float16 qw_s[64 * 68];      // q + r_w_bias (f16)
  __shared__ _Float16 delta_s[64];        // r_r_bias - r_w_bias
  __shared__ _Float16 kvr_s[128 * 68];    // phase 1: Rsub[128][d]; phase 2: K|V
  __shared__ float    S_s[64 * 132];      // scores / P-band; cols 64..95 overlay P(f16)
  __shared__ float    mrow[64], lrow[64], arow[64];

  const int tid = threadIdx.x;
  const int lane = tid & 31, wv = tid >> 5;
  const int i0 = blockIdx.x * 64;
  const int b = blockIdx.y / cH, h = blockIdx.y % cH;
  const int tm = wv & 3;
  const int oc0 = (wv >> 2) * 2;

  // load Q tile (+rw bias) vectorized: 1024 float4, 4 per thread
  {
    float4 rq[4];
#pragma unroll
    for (int i = 0; i < 4; ++i) {
      int idx4 = tid + i * 256;
      int u = idx4 >> 4, dd4 = (idx4 & 15) << 2;
      rq[i] = *(const float4*)&qkv[(size_t)((i0 + u) * cB + b) * (3 * cE) +
                                   h * cD + dd4];
    }
#pragma unroll
    for (int i = 0; i < 4; ++i) {
      int idx4 = tid + i * 256;
      int u = idx4 >> 4, dd4 = (idx4 & 15) << 2;
      float4 w = *(const float4*)&rw[h * cD + dd4];
      float4 s;
      s.x = rq[i].x + w.x; s.y = rq[i].y + w.y;
      s.z = rq[i].z + w.z; s.w = rq[i].w + w.w;
      *(v4h*)&qw_s[u * 68 + dd4] = cvt4(s);
    }
  }
  if (tid < 64) {
    delta_s[tid] = (_Float16)(rr[h * cD + tid] - rw[h * cD + tid]);
    mrow[tid] = -INFINITY;
    lrow[tid] = 0.f;
  }

  v8f o0 = {}, o1 = {};
  const int ntiles = (i0 >> 6) + 1;       // causal: j0 <= i0

  for (int t = 0; t < ntiles; ++t) {
    const int j0 = t * 64;
    __syncthreads();                       // kvr_s reuse fence (and first-iter init)

    // ---- stage Rsub rows [base .. base+127]: 2048 float4, 8 per thread
    {
      const int base = cT - 64 - i0 + j0;  // >= 0 for j0 <= i0
      float4 rv[8];
#pragma unroll
      for (int i = 0; i < 8; ++i) {
        int idx4 = tid + i * 256;
        int c = idx4 >> 4, dd4 = (idx4 & 15) << 2;
        int m = base + c;
        if (m > cT - 1) m = cT - 1;        // over-range rows are masked later
        rv[i] = *(const float4*)&rbuf[(size_t)m * cE + h * cD + dd4];
      }
#pragma unroll
      for (int i = 0; i < 8; ++i) {
        int idx4 = tid + i * 256;
        int c = idx4 >> 4, dd4 = (idx4 & 15) << 2;
        *(v4h*)&kvr_s[c * 68 + dd4] = cvt4(rv[i]);
      }
    }
    __syncthreads();

    // ---- P band: S[64][128] = (q + rr) · Rsub^T  (WMMA, 32 tiles / 8 waves)
    {
#pragma unroll
      for (int j = 0; j < 4; ++j) {
        int tn = (wv >> 2) * 4 + j;
        v8f acc = {};
#pragma unroll
        for (int k0 = 0; k0 < 64; k0 += 32) {
          v16h a;
          const int hl = lane >> 4;
          const int m = 16 * tm + (lane & 15);
#pragma unroll
          for (int e = 0; e < 16; ++e) {
            int kk = k0 + ((e >> 3) << 4) + (hl << 3) + (e & 7);
            a[e] = (_Float16)((float)qw_s[m * 68 + kk] + (float)delta_s[kk]);
          }
          v16h bfr = frag_bt(kvr_s, 68, 16 * tn + (lane & 15), lane, k0);
          acc = mma16(a, bfr, acc);
        }
        int col = 16 * tn + (lane & 15);
#pragma unroll
        for (int r = 0; r < 8; ++r)
          S_s[(16 * tm + ((lane >> 4) << 3) + r) * 132 + col] = acc[r];
      }
    }
    __syncthreads();

    // ---- Toeplitz gather BD[u][c] = P[u][63-u+c]; then stage K,V tiles
    {
      int rt = tid >> 2, q4 = tid & 3;     // 4 lanes per row, same wave -> lockstep
      float tmp[16];
      int src0 = 63 - rt + q4 * 16;
#pragma unroll
      for (int ii = 0; ii < 16; ++ii) tmp[ii] = S_s[rt * 132 + src0 + ii];
#pragma unroll
      for (int ii = 0; ii < 16; ++ii) S_s[rt * 132 + q4 * 16 + ii] = tmp[ii];
      // K,V tiles: 2048 float4, 8 per thread
      float4 rv[8];
#pragma unroll
      for (int i = 0; i < 8; ++i) {
        int idx4 = tid + i * 256;
        int which = idx4 >> 10;            // 0 -> K, 1 -> V
        int c = (idx4 >> 4) & 63, dd4 = (idx4 & 15) << 2;
        rv[i] = *(const float4*)&qkv[(size_t)((j0 + c) * cB + b) * (3 * cE) +
                                     (size_t)(which + 1) * cE + h * cD + dd4];
      }
#pragma unroll
      for (int i = 0; i < 8; ++i) {
        int idx4 = tid + i * 256;
        int which = idx4 >> 10;
        int c = (idx4 >> 4) & 63, dd4 = (idx4 & 15) << 2;
        *(v4h*)&kvr_s[(which * 64 + c) * 68 + dd4] = cvt4(rv[i]);
      }
    }
    __syncthreads();

    // ---- AC WMMA with BD as C-init; scale + causal mask; write scores
    {
#pragma unroll
      for (int j = 0; j < 2; ++j) {
        int tn = oc0 + j;
        int col = 16 * tn + (lane & 15);
        v8f acc;
#pragma unroll
        for (int r = 0; r < 8; ++r)
          acc[r] = S_s[(16 * tm + ((lane >> 4) << 3) + r) * 132 + col];
#pragma unroll
        for (int k0 = 0; k0 < 64; k0 += 32) {
          v16h a = frag_a(qw_s, 68, 16 * tm + (lane & 15), lane, k0);
          v16h bfr = frag_bt(kvr_s, 68, 16 * tn + (lane & 15), lane, k0); // K rows
          acc = mma16(a, bfr, acc);
        }
#pragma unroll
        for (int r = 0; r < 8; ++r) {
          int row = 16 * tm + ((lane >> 4) << 3) + r;
          int gi = i0 + row, gj = j0 + col;
          S_s[row * 132 + col] = (gj > gi) ? -1e9f : acc[r] * cSCALE;
        }
      }
    }
    __syncthreads();

    // ---- online softmax over this tile; P -> f16 overlay at cols [64..95]
    {
      int row = tid >> 2, q4 = tid & 3;
      float mold = mrow[row];
      float mx = -INFINITY;
#pragma unroll
      for (int ii = 0; ii < 16; ++ii)
        mx = fmaxf(mx, S_s[row * 132 + q4 * 16 + ii]);
      mx = fmaxf(mx, __shfl_xor(mx, 1, 32));
      mx = fmaxf(mx, __shfl_xor(mx, 2, 32));
      float mnew = fmaxf(mold, mx);
      float alpha = expf(mold - mnew);     // mold=-inf, mnew finite -> 0
      _Float16* prow = (_Float16*)&S_s[row * 132 + 64];
      float sum = 0.f;
#pragma unroll
      for (int ii = 0; ii < 16; ++ii) {
        float p = expf(S_s[row * 132 + q4 * 16 + ii] - mnew);
        prow[q4 * 16 + ii] = (_Float16)p;
        sum += p;
      }
      sum += __shfl_xor(sum, 1, 32);
      sum += __shfl_xor(sum, 2, 32);
      if (q4 == 0) {
        mrow[row] = mnew;
        lrow[row] = lrow[row] * alpha + sum;
        arow[row] = alpha;
      }
    }
    __syncthreads();

    // ---- rescale O and accumulate O += P · V (WMMA)
    {
#pragma unroll
      for (int r = 0; r < 8; ++r) {
        float al = arow[16 * tm + ((lane >> 4) << 3) + r];
        o0[r] *= al;
        o1[r] *= al;
      }
#pragma unroll
      for (int k0 = 0; k0 < 64; k0 += 32) {
        v16h a;
        const int hl = lane >> 4;
        const int m = 16 * tm + (lane & 15);
        const _Float16* prow = (const _Float16*)&S_s[m * 132 + 64];
#pragma unroll
        for (int e = 0; e < 16; ++e) {
          int kk = k0 + ((e >> 3) << 4) + (hl << 3) + (e & 7);
          a[e] = prow[kk];
        }
        v16h b0, b1;
#pragma unroll
        for (int e = 0; e < 16; ++e) {
          int kk = k0 + ((e >> 3) << 4) + (hl << 3) + (e & 7); // key index
          b0[e] = kvr_s[(64 + kk) * 68 + 16 * oc0 + (lane & 15)];        // V[k][dd]
          b1[e] = kvr_s[(64 + kk) * 68 + 16 * (oc0 + 1) + (lane & 15)];
        }
        o0 = mma16(a, b0, o0);
        o1 = mma16(a, b1, o1);
      }
    }
  }
  __syncthreads();

  // ---- normalize and write ctx[t, b, h*D + dd]
  {
    int colA = 16 * oc0 + (lane & 15);
#pragma unroll
    for (int r = 0; r < 8; ++r) {
      int u = 16 * tm + ((lane >> 4) << 3) + r;
      float inv = 1.f / lrow[u];
      size_t dst = (size_t)((i0 + u) * cB + b) * cE + h * cD;
      ctx[dst + colA] = o0[r] * inv;
      ctx[dst + colA + 16] = o1[r] * inv;
    }
  }
}

// ---------------------------------------------------------------------------
extern "C" void kernel_launch(void* const* d_in, const int* in_sizes, int n_in,
                              void* d_out, int out_size, void* d_ws, size_t ws_size,
                              hipStream_t stream) {
  (void)in_sizes; (void)n_in; (void)out_size; (void)ws_size;
  const float* x   = (const float*)d_in[0];
  const float* pos = (const float*)d_in[1];
  // d_in[2] attn_mask unused: causal mask computed analytically in rel_attn.

  // workspace layout (floats)
  float* ws = (float*)d_ws;
  size_t off = 0;
  float* in_w  = ws + off; off += (size_t)3 * cE * cE;
  float* in_b  = ws + off; off += (size_t)3 * cE;
  float* pos_w = ws + off; off += (size_t)cE * cE;
  float* pos_b = ws + off; off += (size_t)cE;
  float* out_w = ws + off; off += (size_t)cE * cE;
  float* out_b = ws + off; off += (size_t)cE;
  float* rwb   = ws + off; off += (size_t)cH * cD;
  float* rrb   = ws + off; off += (size_t)cH * cD;
  float* qkv   = ws + off; off += (size_t)cT * cB * 3 * cE;
  float* rbuf  = ws + off; off += (size_t)cT * cE;
  float* ctxb  = ws + off; off += (size_t)cT * cB * cE;

  auto samp = [&](int i, float* outp, int n) {
    int n4 = n >> 2;
    sample_gauss4<<<(n4 + 255) / 256, 256, 0, stream>>>(
        (const float4*)d_in[i], (const float4*)d_in[i + 1],
        (const float4*)d_in[i + 2], (float4*)outp, n4);
  };
  samp(3,  in_w,  3 * cE * cE);
  samp(6,  in_b,  3 * cE);
  samp(9,  pos_w, cE * cE);
  samp(12, pos_b, cE);
  samp(15, out_w, cE * cE);
  samp(18, out_b, cE);
  samp(21, rwb,   cH * cD);
  samp(24, rrb,   cH * cD);

  // qkv = x @ in_w^T + in_b      [T*B, 3E]
  gemm_xwT<<<dim3(3 * cE / 128, (cT * cB) / 64), 256, 0, stream>>>(
      x, in_w, in_b, qkv, cT * cB, 3 * cE, cE);
  // r = pos @ pos_w^T + pos_b    [T, E]
  gemm_xwT<<<dim3(cE / 128, cT / 64), 256, 0, stream>>>(
      pos, pos_w, pos_b, rbuf, cT, cE, cE);
  // fused relative flash attention -> ctx [T*B, E]
  rel_attn<<<dim3(cT / 64, cB * cH), 256, 0, stream>>>(qkv, rbuf, rwb, rrb, ctxb);
  // out = ctx @ out_w^T + out_b  [T*B, E]
  gemm_xwT<<<dim3(cE / 128, (cT * cB) / 64), 256, 0, stream>>>(
      ctxb, out_w, out_b, (float*)d_out, cT * cB, cE, cE);
}
